// KernelNN_41884521071238
// MI455X (gfx1250) — matrix operand order repaired and checked by reading, not compiled, not run on hardware
//
#include <hip/hip_runtime.h>

// ---------------------------------------------------------------------------
// NNConv GNN for MI455X (gfx1250, wave32, WMMA).
//
// msg[e,:] = (kh[e,:] (x) x[src[e],:]) @ K2r + x[src[e],:] @ Kb
//   -> [E,288] x [288,32] GEMM per layer; A built on the fly in packed f16,
//      B pre-swizzled into per-lane WMMA fragment layout (ds_load_b128).
// Scatter-mean via global f32 atomics; counts precomputed once.
// Node update h@root as a second WMMA kernel (K=32 == one chunk).
// ---------------------------------------------------------------------------

typedef __attribute__((ext_vector_type(16))) _Float16 v16h;
typedef __attribute__((ext_vector_type(8)))  float    v8f;

#define NNODES 30000
#define NEDGES 120000
#define WID    32
#define KEXT   288                 // 256 (kh (x) x) + 32 (bias block)
#define NCHUNK 9                   // KEXT / 32
#define NFRAG  (NCHUNK * 2)        // 18 B-fragments (9 K-chunks x 2 N-tiles)
#define EDGE_TILES (NEDGES / 16)   // 7500
#define NODE_TILES (NNODES / 16)   // 1875
#define WPB 8                      // waves per block (256 threads)

// ---------------- small elementwise kernels --------------------------------

__global__ void fc1_kernel(const float* __restrict__ x,
                           const float* __restrict__ w,
                           const float* __restrict__ b,
                           float* __restrict__ h) {
  int idx = blockIdx.x * blockDim.x + threadIdx.x;
  if (idx >= NNODES * WID) return;
  int n = idx >> 5, j = idx & 31;
  h[idx] = x[n] * w[j] + b[j];
}

__global__ void edge_mlp_kernel(const float* __restrict__ ea,   // [E,6]
                                const float* __restrict__ w1,   // [6,8]
                                const float* __restrict__ b1,   // [8]
                                float* __restrict__ kh) {       // [E,8]
  int e = blockIdx.x * blockDim.x + threadIdx.x;
  if (e >= NEDGES) return;
  float a[6];
#pragma unroll
  for (int i = 0; i < 6; ++i) a[i] = ea[e * 6 + i];
#pragma unroll
  for (int hh = 0; hh < 8; ++hh) {
    float s = b1[hh];
#pragma unroll
    for (int i = 0; i < 6; ++i) s += a[i] * w1[i * 8 + hh];
    kh[e * 8 + hh] = fmaxf(s, 0.0f);
  }
}

// Pack K2r||Kb directly into per-lane WMMA B-fragment layout:
//   frag f = c*2 + nt  (c = K-chunk, nt = N-tile)
//   lane l, element t: k = c*32 + 8*(l>>4) + (t&7) + 16*(t>>3) ; n = nt*16 + (l&15)
//   Bfrag[f*32 + l][t] = B[k][n]
//   B[k][n]: k<256 -> k_w2[(k>>5)*1024 + (k&31)*32 + n] ; else k_b2[(k-256)*32+n]
__global__ void pack_b_kernel(const float* __restrict__ k_w2,
                              const float* __restrict__ k_b2,
                              _Float16* __restrict__ Bfrag) {   // [18*32][16]
  int idx = blockIdx.x * blockDim.x + threadIdx.x;
  if (idx >= NFRAG * 32 * 16) return;
  int t = idx & 15;
  int l = (idx >> 4) & 31;
  int f = idx >> 9;
  int c = f >> 1, nt = f & 1;
  int k = c * 32 + ((l >> 4) << 3) + (t & 7) + ((t >> 3) << 4);
  int n = nt * 16 + (l & 15);
  float v;
  if (k < 256) v = k_w2[(k >> 5) * 1024 + (k & 31) * 32 + n];
  else         v = k_b2[(k - 256) * 32 + n];
  Bfrag[idx] = (_Float16)v;
}

// Same fragment packing for root [32,32]: 2 fragments (nt = 0,1), K-chunk 0.
__global__ void pack_root_kernel(const float* __restrict__ root,
                                 _Float16* __restrict__ Rfrag) { // [2*32][16]
  int idx = blockIdx.x * blockDim.x + threadIdx.x;
  if (idx >= 2 * 32 * 16) return;
  int t = idx & 15;
  int l = (idx >> 4) & 31;
  int nt = idx >> 9;
  int k = ((l >> 4) << 3) + (t & 7) + ((t >> 3) << 4);
  int n = nt * 16 + (l & 15);
  Rfrag[idx] = (_Float16)root[k * 32 + n];
}

__global__ void zero_i32_kernel(int* __restrict__ p, int n) {
  int idx = blockIdx.x * blockDim.x + threadIdx.x;
  if (idx < n) p[idx] = 0;
}

__global__ void zero_f32_kernel(float* __restrict__ p, int n) {
  int idx = blockIdx.x * blockDim.x + threadIdx.x;
  if (idx < n) p[idx] = 0.0f;
}

__global__ void count_kernel(const int* __restrict__ dst, int* __restrict__ cnt) {
  int e = blockIdx.x * blockDim.x + threadIdx.x;
  if (e >= NEDGES) return;
  atomicAdd(&cnt[dst[e]], 1);
}

__global__ void inv_kernel(const int* __restrict__ cnt, float* __restrict__ inv) {
  int n = blockIdx.x * blockDim.x + threadIdx.x;
  if (n >= NNODES) return;
  inv[n] = 1.0f / fmaxf((float)cnt[n], 1.0f);
}

// ---------------- message GEMM + scatter (WMMA) ----------------------------
// One wave handles 16 edges: D[16x32] = A[16x288] x B[288x32].
// Chunk c<8: A = kh[:,c] * xrow (packed f16) ; chunk 8: A = xrow (bias block).

__global__ __launch_bounds__(256)
void msg_kernel(const float* __restrict__ hin,   // [N,32]
                const float* __restrict__ kh,    // [E,8]
                const v16h* __restrict__ Bfrag,  // [18*32] fragments
                const int* __restrict__ src,
                const int* __restrict__ dst,
                float* __restrict__ agg) {       // [N,32]
  __shared__ v16h  Blds[NFRAG * 32];             // 18 KB
  __shared__ float xs[WPB][16][WID];             // 16 KB

  const int tid = threadIdx.x;
  // stage B fragments (straight copy, whole block)
  for (int i = tid; i < NFRAG * 32; i += 256) Blds[i] = Bfrag[i];

  const int wave = tid >> 5, lane = tid & 31;
  const int tile = blockIdx.x * WPB + wave;
  const int e0 = tile * 16;

  if (tile < EDGE_TILES) {
#pragma unroll 4
    for (int r = 0; r < 16; ++r) {
      int s = src[e0 + r];                  // uniform across the wave
      xs[wave][r][lane] = hin[s * WID + lane];
    }
  }
  __syncthreads();
  if (tile >= EDGE_TILES) return;

  const int m    = lane & 15;               // A row (edge within tile)
  const int koff = (lane >> 4) * 8;         // A K sub-offset per ISA layout
  const int n    = lane & 15;               // D column (within N-tile)

  // hoist this lane's 16 x elements (chunk-invariant): koff..koff+7, koff+16..+23
  // and convert to f16 once (v16h xh); per-chunk A-build is pure v_pk_mul_f16.
  v16h xh;
  {
    float xv[16];
    float4* xvv = (float4*)xv;
    const float4* xr = (const float4*)xs[wave][m];
    int q = koff >> 2;                      // 0 or 2
    xvv[0] = xr[q];
    xvv[1] = xr[q + 1];
    xvv[2] = xr[q + 4];
    xvv[3] = xr[q + 5];
#pragma unroll
    for (int t = 0; t < 16; ++t) xh[t] = (_Float16)xv[t];
  }
  // hoist this row's 8 kh values (f16)
  _Float16 khh[8];
  {
    float khr[8];
    float4* kv = (float4*)khr;
    const float4* kp = (const float4*)&kh[(e0 + m) * 8];
    kv[0] = kp[0];
    kv[1] = kp[1];
#pragma unroll
    for (int c = 0; c < 8; ++c) khh[c] = (_Float16)khr[c];
  }

  v8f acc0 = {}, acc1 = {};
#pragma unroll
  for (int c = 0; c < NCHUNK; ++c) {
    v16h a = (c < 8) ? (xh * khh[c]) : xh;  // vector*scalar splat -> v_pk_mul_f16
    v16h b0 = Blds[(c * 2 + 0) * 32 + lane];
    v16h b1 = Blds[(c * 2 + 1) * 32 + lane];
    acc0 = __builtin_amdgcn_wmma_f32_16x16x32_f16(false, a, false, b0,
                                                  (short)0, acc0, false, false);
    acc1 = __builtin_amdgcn_wmma_f32_16x16x32_f16(false, a, false, b1,
                                                  (short)0, acc1, false, false);
  }

  // scatter: vgpr r holds row M = r + 8*(lane>=16), columns n / n+16
#pragma unroll
  for (int r = 0; r < 8; ++r) {
    int mm = r + ((lane >> 4) << 3);
    int d  = dst[e0 + mm];
    unsafeAtomicAdd(&agg[d * WID + n],      acc0[r]);
    unsafeAtomicAdd(&agg[d * WID + 16 + n], acc1[r]);
  }
}

// ---------------- node update: relu(agg*inv + h@root + bias) ---------------

__global__ __launch_bounds__(256)
void node_kernel(const float* __restrict__ hin,
                 const float* __restrict__ agg,
                 const float* __restrict__ inv,
                 const v16h* __restrict__ Rfrag,  // [2*32] fragments
                 const float* __restrict__ bias,  // [32]
                 float* __restrict__ hout) {
  __shared__ v16h  Rlds[2 * 32];            // 2 KB
  __shared__ float xs[WPB][16][WID];        // 16 KB

  const int tid = threadIdx.x;
  for (int i = tid; i < 2 * 32; i += 256) Rlds[i] = Rfrag[i];

  const int wave = tid >> 5, lane = tid & 31;
  const int tile = blockIdx.x * WPB + wave;
  const int n0 = tile * 16;

  if (tile < NODE_TILES) {
#pragma unroll 4
    for (int r = 0; r < 16; ++r)
      xs[wave][r][lane] = hin[(n0 + r) * WID + lane];
  }
  __syncthreads();
  if (tile >= NODE_TILES) return;

  const int m    = lane & 15;
  const int koff = (lane >> 4) * 8;
  const int n    = lane & 15;

  v16h a;
  {
    float xv[16];
    float4* xvv = (float4*)xv;
    const float4* xr = (const float4*)xs[wave][m];
    int q = koff >> 2;
    xvv[0] = xr[q];
    xvv[1] = xr[q + 1];
    xvv[2] = xr[q + 4];
    xvv[3] = xr[q + 5];
#pragma unroll
    for (int t = 0; t < 16; ++t) a[t] = (_Float16)xv[t];
  }
  v16h b0 = Rlds[lane];
  v16h b1 = Rlds[32 + lane];

  v8f acc0 = {}, acc1 = {};
  acc0 = __builtin_amdgcn_wmma_f32_16x16x32_f16(false, a, false, b0,
                                                (short)0, acc0, false, false);
  acc1 = __builtin_amdgcn_wmma_f32_16x16x32_f16(false, a, false, b1,
                                                (short)0, acc1, false, false);

#pragma unroll
  for (int r = 0; r < 8; ++r) {
    int mm   = r + ((lane >> 4) << 3);
    int node = n0 + mm;
    float iv = inv[node];
    float v0 = acc0[r] + agg[node * WID + n] * iv + bias[n];
    float v1 = acc1[r] + agg[node * WID + 16 + n] * iv + bias[16 + n];
    hout[node * WID + n]      = fmaxf(v0, 0.0f);
    hout[node * WID + 16 + n] = fmaxf(v1, 0.0f);
  }
}

// ---------------- fc2: [N,32] @ [32,1] + b ---------------------------------

__global__ __launch_bounds__(256)
void fc2_kernel(const float* __restrict__ h,
                const float* __restrict__ w,
                const float* __restrict__ b,
                float* __restrict__ out) {
  int gwave = (blockIdx.x * blockDim.x + threadIdx.x) >> 5;
  int lane = threadIdx.x & 31;
  if (gwave >= NNODES) return;
  float v = h[gwave * WID + lane] * w[lane];
#pragma unroll
  for (int off = 16; off > 0; off >>= 1) v += __shfl_xor(v, off, 32);
  if (lane == 0) out[gwave] = v + b[0];
}

// ---------------- host driver ----------------------------------------------

extern "C" void kernel_launch(void* const* d_in, const int* in_sizes, int n_in,
                              void* d_out, int out_size, void* d_ws, size_t ws_size,
                              hipStream_t stream) {
  const float* x      = (const float*)d_in[0];
  const int*   src    = (const int*)d_in[1];            // edge_index row 0 (int32)
  const int*   dst    = ((const int*)d_in[1]) + NEDGES; // edge_index row 1
  const float* ea     = (const float*)d_in[2];
  const float* fc1_w  = (const float*)d_in[3];
  const float* fc1_b  = (const float*)d_in[4];
  const float* k_w1   = (const float*)d_in[5];
  const float* k_b1   = (const float*)d_in[6];
  const float* k_w2   = (const float*)d_in[7];
  const float* k_b2   = (const float*)d_in[8];
  const float* root   = (const float*)d_in[9];
  const float* cbias  = (const float*)d_in[10];
  const float* fc2_w  = (const float*)d_in[11];
  const float* fc2_b  = (const float*)d_in[12];
  float* out = (float*)d_out;

  // workspace layout (bytes)
  char* ws = (char*)d_ws;
  const size_t HSZ = (size_t)NNODES * WID * sizeof(float);   // 3.84 MB
  float*     hA    = (float*)(ws);
  float*     hB    = (float*)(ws + HSZ);
  float*     agg   = (float*)(ws + 2 * HSZ);
  float*     kh    = (float*)(ws + 3 * HSZ);                 // E*8 f32
  int*       cnt   = (int*)  (ws + 4 * HSZ);
  float*     inv   = (float*)(ws + 4 * HSZ + 128 * 1024);
  _Float16*  Bh    = (_Float16*)(ws + 4 * HSZ + 256 * 1024); // 18 KB fragments
  _Float16*  rooth = (_Float16*)(ws + 4 * HSZ + 256 * 1024 + 32 * 1024);

  const int T = 256;
  // prep
  fc1_kernel<<<(NNODES * WID + T - 1) / T, T, 0, stream>>>(x, fc1_w, fc1_b, hA);
  edge_mlp_kernel<<<(NEDGES + T - 1) / T, T, 0, stream>>>(ea, k_w1, k_b1, kh);
  pack_b_kernel<<<(NFRAG * 32 * 16 + T - 1) / T, T, 0, stream>>>(k_w2, k_b2, Bh);
  pack_root_kernel<<<(2 * 32 * 16 + T - 1) / T, T, 0, stream>>>(root, rooth);
  zero_i32_kernel<<<(NNODES + T - 1) / T, T, 0, stream>>>(cnt, NNODES);
  count_kernel<<<(NEDGES + T - 1) / T, T, 0, stream>>>(dst, cnt);
  inv_kernel<<<(NNODES + T - 1) / T, T, 0, stream>>>(cnt, inv);

  const int msg_blocks  = (EDGE_TILES + WPB - 1) / WPB;   // 938
  const int node_blocks = (NODE_TILES + WPB - 1) / WPB;   // 235

  float* hi = hA;
  float* ho = hB;
  for (int layer = 0; layer < 4; ++layer) {
    zero_f32_kernel<<<(NNODES * WID + T - 1) / T, T, 0, stream>>>(agg, NNODES * WID);
    msg_kernel<<<msg_blocks, T, 0, stream>>>(hi, kh, (const v16h*)Bh, src, dst, agg);
    node_kernel<<<node_blocks, T, 0, stream>>>(hi, agg, inv, (const v16h*)rooth,
                                               cbias, ho);
    float* tmp = hi; hi = ho; ho = tmp;
  }
  // after 4 swaps, result is in hi (== hA)
  fc2_kernel<<<(NNODES * 32 + T - 1) / T, T, 0, stream>>>(hi, fc2_w, fc2_b, out);
}
